// ConvBlock_49924699849381
// MI455X (gfx1250) — compile-verified, matrix operand order checked by reading, not run on hardware
//
#include <hip/hip_runtime.h>
#include <cstdint>

// ---------------------------------------------------------------------------
// ONN ConvBlock for MI455X (gfx1250):
//   prep:  MRR phase -> transmission, bf16 weight matrix [256][1152] (K
//          reordered as (kh*3+kw)*128+ic), plus folded BN scale/bias.
//   conv:  implicit-GEMM, V_WMMA_F32_16X16X32_BF16, f32 accumulation,
//          double-buffered LDS tiles, TDM weight-tile DMA (when available),
//          BN+ReLU6 epilogue.
// ---------------------------------------------------------------------------

typedef __attribute__((ext_vector_type(16))) __bf16 v16bf;
typedef __attribute__((ext_vector_type(8)))  __bf16 v8bf;
typedef __attribute__((ext_vector_type(8)))  float  v8f;
typedef __attribute__((ext_vector_type(4)))  unsigned int u32x4;
typedef __attribute__((ext_vector_type(8)))  int i32x8;
typedef __attribute__((ext_vector_type(4)))  int i32x4;

#define IN_CH   128
#define OUT_CH  256
#define NIMG    64
#define IMG_H   56
#define IMG_W   56
#define HW      3136          // 56*56
#define KTOT    1152          // 128*9
#define KSTEP   32
#define KSTEPS  36            // 1152/32
#define BLK_OC  128
#define BLK_PX  128
#define LDSR    40            // shorts per LDS row: 32 data + 8 pad (80B)

#if defined(__gfx1250__) && __has_builtin(__builtin_amdgcn_tensor_load_to_lds) \
                         && __has_builtin(__builtin_amdgcn_s_wait_tensorcnt)
#define HAVE_TDM 1
#endif

__device__ __forceinline__ unsigned short f2bf(float f) {
  unsigned int u = __float_as_uint(f);
  u += 0x7FFFu + ((u >> 16) & 1u);       // round-to-nearest-even
  return (unsigned short)(u >> 16);
}

// Pack two f32 -> packed bf16 pair (a in [15:0], b in [31:16]).
// round-half-up via +0x8000, then one v_perm_b32 grabs both high halves.
__device__ __forceinline__ unsigned int pack_bf16(float a, float b) {
#if __has_builtin(__builtin_amdgcn_perm)
  unsigned int ua = __float_as_uint(a) + 0x8000u;
  unsigned int ub = __float_as_uint(b) + 0x8000u;
  return __builtin_amdgcn_perm(ub, ua, 0x07060302u); // [b3 b2 a3 a2]
#else
  return (unsigned int)f2bf(a) | ((unsigned int)f2bf(b) << 16);
#endif
}

// Build one 16x32 bf16 WMMA operand from an LDS tile stored row-major
// ([row][32] shorts, row stride LDSR). Matches the CDNA5 16-bit A/B layout:
//   lane<16  : row = lane,    K = 0..7 and 16..23
//   lane>=16 : row = lane-16, K = 8..15 and 24..31
__device__ __forceinline__ v16bf frag_from_lds(const unsigned short* base,
                                               int row0, int lane) {
  const unsigned short* p =
      base + (row0 + (lane & 15)) * LDSR + ((lane & 16) ? 8 : 0);
  union { v16bf v; v8bf h[2]; } u;
  u.h[0] = *(const v8bf*)(p);        // ds_read_b128
  u.h[1] = *(const v8bf*)(p + 16);   // ds_read_b128
  return u.v;
}

#ifdef HAVE_TDM
// DMA one 32(k) x 128(oc) bf16 weight tile (row stride KTOT shorts) into LDS,
// inserting 16B of padding after every 64B row -> the 80B LDSR layout.
__device__ __forceinline__ void tdm_load_wtile(const unsigned short* gsrc,
                                               unsigned int lds_byte_addr) {
  unsigned long long ga = (unsigned long long)(uintptr_t)gsrc;
  u32x4 g0;
  g0[0] = 1u;                                       // count=1, user D#
  g0[1] = lds_byte_addr;                            // lds_addr
  g0[2] = (unsigned int)ga;                         // global_addr[31:0]
  g0[3] = (unsigned int)((ga >> 32) & 0x01FFFFFFull) | (2u << 30); // type=2
  i32x8 g1;
  g1[0] = (1 << 16)        // data_size = 2 bytes
        | (1 << 20)        // pad_enable
        | (3 << 22)        // pad_interval: 16 DWORDs (64B) between pads
        | (3 << 25);       // pad_amount  : 4 DWORDs (16B)
  g1[1] = (int)((KTOT & 0xFFFF) << 16);             // tensor_dim0 lo16
  g1[2] = (int)((KTOT >> 16) | (128u << 16));       // dim0 hi | tensor_dim1 lo
  g1[3] = (int)(32u << 16);                         // dim1 hi | tile_dim0 = 32
  g1[4] = 128;                                      // tile_dim1=128, tile_dim2=0
  g1[5] = KTOT;                                     // tensor_dim0_stride lo32
  g1[6] = 0;
  g1[7] = 0;
  i32x4 gz = {0, 0, 0, 0};
#if __clang_major__ >= 23
  i32x8 gz8 = {0, 0, 0, 0, 0, 0, 0, 0};
  __builtin_amdgcn_tensor_load_to_lds(g0, g1, gz, gz, gz8, 0);
#else
  __builtin_amdgcn_tensor_load_to_lds(g0, g1, gz, gz, 0);
#endif
}
#else
__device__ __forceinline__ void stage_a_manual(unsigned short* dst,
                                               const unsigned short* wrow,
                                               int tid) {
  const int row  = tid >> 1;              // 128 rows, 2 threads/row
  const int koff = (tid & 1) * 16;
  const unsigned short* src = wrow + (size_t)row * KTOT + koff;
  uint4 w0 = *(const uint4*)src;
  uint4 w1 = *(const uint4*)(src + 8);
  *(uint4*)&dst[row * LDSR + koff]     = w0;
  *(uint4*)&dst[row * LDSR + koff + 8] = w1;
}
#endif

// Stage the im2col activation tile: this thread owns one pixel (bpx) and 16
// consecutive input channels (bic0..bic0+15). Pixel coords are pre-decomposed;
// the 16 loads are lane-coalesced b32s with immediate offsets (stride HW*4B).
__device__ __forceinline__ void stage_b(unsigned short* dst, const float* xpix,
                                        int ph, int pw, int bpx, int bic0,
                                        int kb) {
  const int pos = kb >> 2;                    // kh*3+kw (fixed per 4 k-steps)
  const int dh  = pos / 3 - 1;
  const int dw  = pos - (pos / 3) * 3 - 1;
  const int icb = (kb & 3) * 32 + bic0;
  const bool vld = ((unsigned)(ph + dh) < IMG_H) && ((unsigned)(pw + dw) < IMG_W);
  const float* src = xpix + (size_t)icb * HW + dh * IMG_W + dw;
  float v[16];
#pragma unroll
  for (int j = 0; j < 16; ++j) v[j] = 0.0f;
  if (vld) {
#pragma unroll
    for (int j = 0; j < 16; ++j) v[j] = src[(size_t)j * HW];
  }
  unsigned int pk[8];
#pragma unroll
  for (int j = 0; j < 8; ++j) pk[j] = pack_bf16(v[2 * j], v[2 * j + 1]);
  uint4 q0 = {pk[0], pk[1], pk[2], pk[3]};
  uint4 q1 = {pk[4], pk[5], pk[6], pk[7]};
  *(uint4*)&dst[bpx * LDSR + bic0]     = q0;   // ds_write_b128
  *(uint4*)&dst[bpx * LDSR + bic0 + 8] = q1;   // ds_write_b128
}

// ---------------------------------------------------------------------------
__global__ __launch_bounds__(256)
void onn_prep(const float* __restrict__ phase,
              const float* __restrict__ gamma,
              const float* __restrict__ beta,
              const float* __restrict__ rmean,
              const float* __restrict__ rvar,
              unsigned short* __restrict__ wq,
              float* __restrict__ scale,
              float* __restrict__ bias) {
  int idx = blockIdx.x * 256 + threadIdx.x;
  if (idx < OUT_CH * KTOT) {
    int oc  = idx / KTOT;
    int kn  = idx - oc * KTOT;       // new K: (kh*3+kw)*128 + ic
    int ic  = kn & (IN_CH - 1);
    int pos = kn >> 7;               // kh*3+kw
    int kold = ic * 9 + pos;         // original K = ic*9 + kh*3 + kw
    int gy = oc >> 3,   ry = oc & 7;
    int gx = kold >> 3, rx = kold & 7;
    float phi = phase[((gy * 144 + gx) * 8 + ry) * 8 + rx];
    const float a = 0.987f, r = 0.99f, ar = 0.987f * 0.99f;
    float c   = cosf(phi);
    float num = a * a - 2.0f * ar * c + r * r;
    float den = 1.0f  - 2.0f * ar * c + ar * ar;
    wq[idx] = f2bf(num / den);
  }
  if (blockIdx.x == 0 && threadIdx.x < OUT_CH) {
    int oc = threadIdx.x;
    float inv = gamma[oc] * rsqrtf(rvar[oc] + 1e-5f);
    scale[oc] = inv;
    bias[oc]  = beta[oc] - rmean[oc] * inv;
  }
}

// ---------------------------------------------------------------------------
__global__ __launch_bounds__(256)
void onn_conv_wmma(const float* __restrict__ x,
                   const unsigned short* __restrict__ wq,
                   const float* __restrict__ scale,
                   const float* __restrict__ bias,
                   float* __restrict__ out) {
  __shared__ unsigned short lds_a[2][BLK_OC * LDSR];  // weights: [oc][k]
  __shared__ unsigned short lds_b[2][BLK_PX * LDSR];  // im2col : [px][k]

  const int tid  = threadIdx.x;
  const int lane = tid & 31;
  const int wave = tid >> 5;
  const int wm   = wave & 3;                // 4 waves over oc  (32 each)
  const int wn   = wave >> 2;               // 2 waves over px  (64 each)
  const int m0   = blockIdx.x * BLK_PX;     // pixel tile base
  const int oc0  = blockIdx.y * BLK_OC;     // channel tile base

  // ---- per-thread pixel decomposition for B staging (K-loop invariant) ----
  const int bpx  = tid & 127;               // pixel this thread stages
  const int bic0 = (tid >> 7) * 16;         // first of 16 channels it stages
  const int pm   = m0 + bpx;
  const int pn   = pm / HW;
  const int phw  = pm - pn * HW;
  const int ph   = phw / IMG_W;
  const int pw   = phw - ph * IMG_W;
  const float* xpix = x + (size_t)pn * (IN_CH * HW) + ph * IMG_W + pw;

  const unsigned short* wbase = wq + (size_t)oc0 * KTOT;

  v8f acc[2][4];
#pragma unroll
  for (int mi = 0; mi < 2; ++mi)
#pragma unroll
    for (int ni = 0; ni < 4; ++ni) {
      v8f z = {0.f, 0.f, 0.f, 0.f, 0.f, 0.f, 0.f, 0.f};
      acc[mi][ni] = z;
    }

  // ---- prologue: stage tile 0 into buffer 0 ----
#ifdef HAVE_TDM
  if (wave == 0)
    tdm_load_wtile(wbase, (unsigned int)(uintptr_t)&lds_a[0][0]);
#else
  stage_a_manual(&lds_a[0][0], wbase, tid);
#endif
  stage_b(&lds_b[0][0], xpix, ph, pw, bpx, bic0, 0);
#ifdef HAVE_TDM
  __builtin_amdgcn_s_wait_tensorcnt(0);
#endif
  __syncthreads();

  // ---- main loop: ping-pong, stage k+1 while computing k ----
  for (int kb = 0; kb < KSTEPS; ++kb) {
    const int cur = kb & 1;
    const int nxt = cur ^ 1;
    if (kb + 1 < KSTEPS) {
#ifdef HAVE_TDM
      if (wave == 0)
        tdm_load_wtile(wbase + (size_t)(kb + 1) * KSTEP,
                       (unsigned int)(uintptr_t)&lds_a[nxt][0]);
#else
      stage_a_manual(&lds_a[nxt][0], wbase + (size_t)(kb + 1) * KSTEP, tid);
#endif
      stage_b(&lds_b[nxt][0], xpix, ph, pw, bpx, bic0, kb + 1);
    }

    v16bf afr[2], bfr[4];
#pragma unroll
    for (int mi = 0; mi < 2; ++mi)
      afr[mi] = frag_from_lds(&lds_a[cur][0], wm * 32 + mi * 16, lane);
#pragma unroll
    for (int ni = 0; ni < 4; ++ni)
      bfr[ni] = frag_from_lds(&lds_b[cur][0], wn * 64 + ni * 16, lane);
#pragma unroll
    for (int mi = 0; mi < 2; ++mi)
#pragma unroll
      for (int ni = 0; ni < 4; ++ni)
        acc[mi][ni] = __builtin_amdgcn_wmma_f32_16x16x32_bf16(
            false, afr[mi], false, bfr[ni],
            (short)0, acc[mi][ni], false, false);

#ifdef HAVE_TDM
    __builtin_amdgcn_s_wait_tensorcnt(0);   // TDM data landed (wave 0; others trivial)
#endif
    __syncthreads();                        // writes to nxt visible; cur free
  }

  // ---- epilogue: BN + ReLU6, scatter per C-matrix layout ----
  const int col  = lane & 15;               // pixel within 16
  const int roff = (lane >> 4) * 8;         // oc row offset 0/8
  float sc[2][8], bs[2][8];
#pragma unroll
  for (int mi = 0; mi < 2; ++mi)
#pragma unroll
    for (int r = 0; r < 8; ++r) {
      int oc = oc0 + wm * 32 + mi * 16 + roff + r;
      sc[mi][r] = scale[oc];
      bs[mi][r] = bias[oc];
    }
#pragma unroll
  for (int ni = 0; ni < 4; ++ni) {
    int px = m0 + wn * 64 + ni * 16 + col;
    int n  = px / HW;
    int hw = px - n * HW;
    float* orow = out + (size_t)n * (OUT_CH * HW) + hw;
#pragma unroll
    for (int mi = 0; mi < 2; ++mi) {
#pragma unroll
      for (int r = 0; r < 8; ++r) {
        int oc  = oc0 + wm * 32 + mi * 16 + roff + r;
        float v = acc[mi][ni][r] * sc[mi][r] + bs[mi][r];
        v = fminf(fmaxf(v, 0.0f), 6.0f);
        orow[(size_t)oc * HW] = v;
      }
    }
  }
}

// ---------------------------------------------------------------------------
extern "C" void kernel_launch(void* const* d_in, const int* in_sizes, int n_in,
                              void* d_out, int out_size, void* d_ws,
                              size_t ws_size, hipStream_t stream) {
  (void)in_sizes; (void)n_in; (void)out_size; (void)ws_size;
  const float* x     = (const float*)d_in[0];
  const float* phase = (const float*)d_in[1];
  const float* gamma = (const float*)d_in[2];
  const float* beta  = (const float*)d_in[3];
  const float* rmean = (const float*)d_in[4];
  const float* rvar  = (const float*)d_in[5];

  unsigned short* wq = (unsigned short*)d_ws;                  // 256*1152 bf16
  float* scale = (float*)((char*)d_ws +
                          (size_t)OUT_CH * KTOT * sizeof(unsigned short));
  float* bias  = scale + OUT_CH;

  onn_prep<<<(OUT_CH * KTOT + 255) / 256, 256, 0, stream>>>(
      phase, gamma, beta, rmean, rvar, wq, scale, bias);

  dim3 grid((NIMG * HW) / BLK_PX, OUT_CH / BLK_OC);            // 1568 x 2
  onn_conv_wmma<<<grid, 256, 0, stream>>>(x, wq, scale, bias, (float*)d_out);
}
